// SOA_76158360092784
// MI455X (gfx1250) — matrix operand order repaired and checked by reading, not compiled, not run on hardware
//
#include <hip/hip_runtime.h>
#include <hip/hip_bf16.h>

typedef __attribute__((ext_vector_type(16))) _Float16 v16h;
typedef __attribute__((ext_vector_type(8)))  _Float16 v8h;
typedef __attribute__((ext_vector_type(8)))  float    v8f;

#define B_  8
#define N_  2048
#define C_  256
#define DK_ 64
#define DV_ 256
#define BN_ (B_*N_)     /* 16384 */
#define NN_ (N_*N_)     /* 4194304 */

// Pointer types matching the async-load builtin's signature:
//   param0: int4 AS(1)*   (global source)   param1: int4 AS(3)* (LDS dest)
typedef int v4i __attribute__((vector_size(16)));
typedef v4i __attribute__((address_space(1))) g_v4i;
typedef v4i __attribute__((address_space(3))) l_v4i;

__device__ __forceinline__ void async_cp_b128(const _Float16* g, _Float16* l) {
#if __has_builtin(__builtin_amdgcn_global_load_async_to_lds_b128)
  __builtin_amdgcn_global_load_async_to_lds_b128((g_v4i*)g, (l_v4i*)l, 0, 0);
#else
  *(v8h*)l = *(const v8h*)g;     // fallback: plain 16B LDS store
#endif
}
__device__ __forceinline__ void wait_async0() {
#if __has_builtin(__builtin_amdgcn_s_wait_asynccnt)
  __builtin_amdgcn_s_wait_asynccnt(0);
#else
  asm volatile("s_wait_asynccnt 0x0" ::: "memory");
#endif
}

// ---------------- K1: q,k,v projections (fp32 in -> f16 out, WMMA f16/f32acc)
// 1024 M-tiles (over B*N) x 24 out-dim tiles (4 q + 4 k + 16 v), 1 tile/wave.
__global__ __launch_bounds__(256) void proj_kernel(
    const float* __restrict__ x,  const float* __restrict__ Wq,
    const float* __restrict__ Wk, const float* __restrict__ Wv,
    _Float16* __restrict__ qh, _Float16* __restrict__ kh,
    _Float16* __restrict__ vT)
{
  const int lane = threadIdx.x & 31;
  const int tile = __builtin_amdgcn_readfirstlane(blockIdx.x * 8 + (threadIdx.x >> 5));
  const int nt = tile % 24;                    // scalar
  const int mt = tile / 24;                    // scalar
  const int l15 = lane & 15, hs = lane >> 4, kbase = hs * 8;
  const int row = mt * 16 + l15;

  const float* W; int dglob;
  if (nt < 4)      { W = Wq; dglob = nt * 16 + l15; }
  else if (nt < 8) { W = Wk; dglob = (nt - 4) * 16 + l15; }
  else             { W = Wv; dglob = (nt - 8) * 16 + l15; }

  typedef __attribute__((ext_vector_type(8))) float vf8;
  v8f c = {};
  for (int c0 = 0; c0 < C_; c0 += 32) {
    vf8 alo = *(const vf8*)(x + row * C_ + c0 + kbase);        // K = kbase..+7
    vf8 ahi = *(const vf8*)(x + row * C_ + c0 + 16 + kbase);   // K = 16+kbase..
    vf8 blo = *(const vf8*)(W + dglob * C_ + c0 + kbase);
    vf8 bhi = *(const vf8*)(W + dglob * C_ + c0 + 16 + kbase);
    v16h a, bb;
    #pragma unroll
    for (int e = 0; e < 8; ++e) {
      a[e]      = (_Float16)alo[e];
      a[e + 8]  = (_Float16)ahi[e];
      bb[e]     = (_Float16)blo[e];
      bb[e + 8] = (_Float16)bhi[e];
    }
    c = __builtin_amdgcn_wmma_f32_16x16x32_f16(false, a, false, bb,
                                               (short)0, c, false, false);
  }
  #pragma unroll
  for (int j = 0; j < 8; ++j) {
    int rj = mt * 16 + j + 8 * hs;
    _Float16 hv = (_Float16)c[j];
    if (nt < 4)      qh[rj * DK_ + dglob] = hv;
    else if (nt < 8) kh[rj * DK_ + dglob] = hv;
    else {                                     // v stored TRANSPOSED [B][DV][N]
      int bb_ = rj >> 11;
      int n   = rj & (N_ - 1);
      vT[(size_t)bb_ * DV_ * N_ + (size_t)dglob * N_ + n] = hv;
    }
  }
}

// ---------------- K2: E[b,n,m] = q[b,n,:] . k[b,m,:]  (f16 WMMA, K=64)
__global__ __launch_bounds__(256) void energy_kernel(
    const _Float16* __restrict__ qh, const _Float16* __restrict__ kh,
    _Float16* __restrict__ E)
{
  const int lane = threadIdx.x & 31;
  const int wave = __builtin_amdgcn_readfirstlane(blockIdx.x * 8 + (threadIdx.x >> 5));
  const int b = wave >> 14;
  const int rem = wave & 16383;
  const int ntile = rem >> 7;
  const int mtile = rem & 127;
  const int l15 = lane & 15, hs = lane >> 4, kbase = hs * 8;
  const _Float16* qrow = qh + (size_t)(b * N_ + ntile * 16 + l15) * DK_;
  const _Float16* krow = kh + (size_t)(b * N_ + mtile * 16 + l15) * DK_;

  v8f c = {};
  #pragma unroll
  for (int d0 = 0; d0 < DK_; d0 += 32) {
    v8h alo = *(const v8h*)(qrow + d0 + kbase);
    v8h ahi = *(const v8h*)(qrow + d0 + 16 + kbase);
    v16h a  = __builtin_shufflevector(alo, ahi, 0,1,2,3,4,5,6,7,8,9,10,11,12,13,14,15);
    v8h blo = *(const v8h*)(krow + d0 + kbase);
    v8h bhi = *(const v8h*)(krow + d0 + 16 + kbase);
    v16h bb = __builtin_shufflevector(blo, bhi, 0,1,2,3,4,5,6,7,8,9,10,11,12,13,14,15);
    c = __builtin_amdgcn_wmma_f32_16x16x32_f16(false, a, false, bb,
                                               (short)0, c, false, false);
  }
  size_t base = (size_t)b * NN_;
  #pragma unroll
  for (int j = 0; j < 8; ++j) {
    int n = ntile * 16 + j + 8 * hs;
    E[base + (size_t)n * N_ + mtile * 16 + l15] = (_Float16)c[j];
  }
}

// ---------------- K3a: partial column max/sum-of-exp over 256-row chunks
__global__ __launch_bounds__(256) void colstats_part_kernel(
    const _Float16* __restrict__ E, float* __restrict__ pmax, float* __restrict__ psum)
{
  int cg    = blockIdx.x & 63;                 // 64 column groups of 256
  int chunk = blockIdx.x >> 6;                 // 8 row chunks of 256
  int g = cg * 256 + threadIdx.x;              // [0, B*N)
  int b = g >> 11, m = g & (N_ - 1);
  const _Float16* col = E + (size_t)b * NN_ + m;
  float mx = -3.0e38f, s = 0.f;
  for (int n = chunk * 256; n < chunk * 256 + 256; ++n) {
    float e = (float)col[(size_t)n * N_];
    float nm = fmaxf(mx, e);
    s = s * __expf(mx - nm) + __expf(e - nm);
    mx = nm;
  }
  pmax[chunk * BN_ + g] = mx;
  psum[chunk * BN_ + g] = s;
}

// ---------------- K3b: combine the 8 partials per column
__global__ __launch_bounds__(256) void colstats_comb_kernel(
    const float* __restrict__ pmax, const float* __restrict__ psum,
    float* __restrict__ colmax, float* __restrict__ colsum)
{
  int g = blockIdx.x * 256 + threadIdx.x;      // [0, B*N)
  float mx = -3.0e38f;
  #pragma unroll
  for (int i = 0; i < 8; ++i) mx = fmaxf(mx, pmax[i * BN_ + g]);
  float s = 0.f;
  #pragma unroll
  for (int i = 0; i < 8; ++i) s += psum[i * BN_ + g] * __expf(pmax[i * BN_ + g] - mx);
  colmax[g] = mx;
  colsum[g] = s;
}

// ---------------- K4: P = exp(E - colmax)/colsum (in place) + row sums
__global__ __launch_bounds__(256) void pnorm_kernel(
    _Float16* __restrict__ E, const float* __restrict__ colmax,
    const float* __restrict__ colsum, float* __restrict__ rowsum)
{
  __shared__ float red[256];
  int blk = blockIdx.x;                        // [0, B*N): one row each
  int b = blk >> 11, n = blk & (N_ - 1);
  _Float16* rowp = E + (size_t)b * NN_ + (size_t)n * N_;
  const float* cm = colmax + b * N_;
  const float* cs = colsum + b * N_;
  float s = 0.f;
  #pragma unroll
  for (int i = 0; i < N_ / 256; ++i) {
    int m = threadIdx.x + i * 256;
    float p = __expf((float)rowp[m] - cm[m]) / cs[m];
    rowp[m] = (_Float16)p;
    s += p;
  }
  red[threadIdx.x] = s;
  __syncthreads();
  for (int off = 128; off > 0; off >>= 1) {
    if (threadIdx.x < off) red[threadIdx.x] += red[threadIdx.x + off];
    __syncthreads();
  }
  if (threadIdx.x == 0) rowsum[blk] = red[0];
}

// ---------------- K5: out = diag(1/(1e-9+rowsum)) * (P @ v)
// Block (8 waves) computes 128 rows x 128 cols; vT k-chunks staged in LDS via
// async global->LDS, double buffered; 8 accumulators/wave share one A frag.
#define AVK 32
__global__ __launch_bounds__(256) void av_kernel(
    const _Float16* __restrict__ P, const _Float16* __restrict__ vT,
    const float* __restrict__ rowsum, float* __restrict__ out)
{
  __shared__ __align__(64) _Float16 tile[2][128 * AVK];   // 2 x 8KB
  const int lane = threadIdx.x & 31;
  const int w    = __builtin_amdgcn_readfirstlane(threadIdx.x >> 5);
  const int bid  = blockIdx.x;                 // 256 blocks
  const int b    = bid >> 5;
  const int rblk = (bid >> 1) & 15;            // 16 row-blocks of 128
  const int dh   = bid & 1;                    // DV half
  const int dbase = dh * 128;
  const int l15 = lane & 15, hs = lane >> 4, kbase = hs * 8;
  const int n0 = rblk * 128 + w * 16;
  const _Float16* prow  = P  + (size_t)b * NN_ + (size_t)(n0 + l15) * N_;
  const _Float16* vbase = vT + (size_t)b * DV_ * N_;

  v8f acc[8];
  #pragma unroll
  for (int i = 0; i < 8; ++i) acc[i] = (v8f){};

  // stage chunk [m0, m0+32) of vT rows [dbase, dbase+128) into tile[buf]
  auto stage = [&](int buf, int m0) {
    #pragma unroll
    for (int i = 0; i < 2; ++i) {
      int idx  = threadIdx.x + i * 256;        // [0, 512) 16B vectors
      int dl   = idx >> 2;                     // 0..127
      int part = idx & 3;                      // 8-half group within chunk
      async_cp_b128(vbase + (size_t)(dbase + dl) * N_ + m0 + part * 8,
                    &tile[buf][dl * AVK + part * 8]);
    }
  };

  stage(0, 0);
  wait_async0();
  __syncthreads();

  int buf = 0;
  for (int m0 = 0; m0 < N_; m0 += AVK) {
    if (m0 + AVK < N_) stage(buf ^ 1, m0 + AVK);
    v8h alo = *(const v8h*)(prow + m0 + kbase);
    v8h ahi = *(const v8h*)(prow + m0 + 16 + kbase);
    v16h a  = __builtin_shufflevector(alo, ahi, 0,1,2,3,4,5,6,7,8,9,10,11,12,13,14,15);
    #pragma unroll
    for (int dt = 0; dt < 8; ++dt) {
      v8h blo = *(const v8h*)&tile[buf][(dt * 16 + l15) * AVK + kbase];
      v8h bhi = *(const v8h*)&tile[buf][(dt * 16 + l15) * AVK + 16 + kbase];
      v16h bb = __builtin_shufflevector(blo, bhi, 0,1,2,3,4,5,6,7,8,9,10,11,12,13,14,15);
      acc[dt] = __builtin_amdgcn_wmma_f32_16x16x32_f16(false, a, false, bb,
                                                       (short)0, acc[dt], false, false);
    }
    wait_async0();
    __syncthreads();
    buf ^= 1;
  }

  float scl[8];
  #pragma unroll
  for (int j = 0; j < 8; ++j)
    scl[j] = 1.0f / (1e-9f + rowsum[b * N_ + n0 + j + 8 * hs]);
  #pragma unroll
  for (int dt = 0; dt < 8; ++dt) {
    int d = dbase + dt * 16 + l15;
    #pragma unroll
    for (int j = 0; j < 8; ++j) {
      int n = n0 + j + 8 * hs;
      out[(size_t)b * N_ * DV_ + (size_t)n * DV_ + d] = acc[dt][j] * scl[j];
    }
  }
}

extern "C" void kernel_launch(void* const* d_in, const int* in_sizes, int n_in,
                              void* d_out, int out_size, void* d_ws, size_t ws_size,
                              hipStream_t stream) {
  const float* x  = (const float*)d_in[0];
  const float* Wq = (const float*)d_in[1];
  const float* Wk = (const float*)d_in[2];
  const float* Wv = (const float*)d_in[3];
  float* out = (float*)d_out;

  char* ws = (char*)d_ws;
  size_t off = 0;
  auto carve = [&](size_t bytes) {
    size_t p = off;
    off += (bytes + 255) & ~(size_t)255;
    return p;
  };
  _Float16* qh     = (_Float16*)(ws + carve((size_t)BN_ * DK_ * 2));
  _Float16* kh     = (_Float16*)(ws + carve((size_t)BN_ * DK_ * 2));
  _Float16* vT     = (_Float16*)(ws + carve((size_t)B_ * DV_ * N_ * 2));
  _Float16* E      = (_Float16*)(ws + carve((size_t)B_ * NN_ * 2));
  float*    pmax   = (float*)(ws + carve((size_t)8 * BN_ * 4));
  float*    psum   = (float*)(ws + carve((size_t)8 * BN_ * 4));
  float*    colmax = (float*)(ws + carve((size_t)BN_ * 4));
  float*    colsum = (float*)(ws + carve((size_t)BN_ * 4));
  float*    rowsum = (float*)(ws + carve((size_t)BN_ * 4));

  proj_kernel         <<<3072,  256, 0, stream>>>(x, Wq, Wk, Wv, qh, kh, vT);
  energy_kernel       <<<16384, 256, 0, stream>>>(qh, kh, E);
  colstats_part_kernel<<<512,   256, 0, stream>>>(E, pmax, psum);
  colstats_comb_kernel<<<64,    256, 0, stream>>>(pmax, psum, colmax, colsum);
  pnorm_kernel        <<<BN_,   256, 0, stream>>>(E, colmax, colsum, rowsum);
  av_kernel           <<<256,   256, 0, stream>>>(E, vT, rowsum, out);
}
